// ScaledDotProductAttention_62560493633919
// MI455X (gfx1250) — matrix operand order repaired and checked by reading, not compiled, not run on hardware
//
#include <hip/hip_runtime.h>
#include <math.h>

// ---------------------------------------------------------------------------
// Scaled dot-product attention for MI455X (gfx1250, wave32, WMMA).
// B=32, S=2048, D=64. Outputs: output [B,S,D] then attn [B,S,S], fp32.
//
// Roofline: 512MB attn write dominates -> ~30us floor @ 23.3 TB/s. To stay
// near it, minimize dynamic instruction count:
//  - v_wmma_f32_16x16x32_bf16 for both matmuls (bf16 in, fp32 accumulate)
//  - 32-query strip per workgroup, scores resident in 256KB of the 320KB
//    WGP LDS; K-tiles feed 4 WMMAs, V-tiles feed 2 WMMAs (amortized staging)
//  - float4 LDS softmax sweeps, u32 mask loads, b128 nontemporal attn stores
//    (keeps the 192MB L2 free for the per-batch 512KB K/V panels)
// ---------------------------------------------------------------------------

typedef __attribute__((ext_vector_type(16))) __bf16 v16bf;
typedef __attribute__((ext_vector_type(8)))  float  v8f;
typedef __attribute__((ext_vector_type(4)))  float  f4;

union V16 { v16bf v; __bf16 e[16]; };
union V8F { v8f  v; float  e[8];  };

#define BATCH    32
#define SEQ      2048
#define DIM      64
#define SQ       32             // query rows per workgroup (2 WMMA M-tiles)
#define NWAVES   8
#define NTHREADS 256
#define KCHUNK   (SEQ / NWAVES) // 256 keys per wave
#define SCALE    0.125f         // 1/temperature = 1/sqrt(64)

__global__ __launch_bounds__(NTHREADS)
void sdpa_wmma_kernel(const float* __restrict__ q,
                      const float* __restrict__ k,
                      const float* __restrict__ v,
                      const unsigned char* __restrict__ mask,
                      float* __restrict__ out,
                      float* __restrict__ attn)
{
    // 32 x 2048 fp32 score/prob strip: 256 KB LDS (CDNA5 WGP has 320 KB).
    __shared__ float sc[SQ][SEQ];

    const int b     = blockIdx.y;
    const int q0    = blockIdx.x * SQ;
    const int tid   = threadIdx.x;
    const int wave  = tid >> 5;
    const int lane  = tid & 31;
    const int lrow  = lane & 15;   // row (A/D) or column (B) index
    const int lhalf = lane >> 4;   // 16-lane half of the wave

    const float* qb = q + ((size_t)b * SEQ + q0) * DIM;
    const float* kb = k + (size_t)b * SEQ * DIM;
    const float* vb = v + (size_t)b * SEQ * DIM;
    const unsigned char* mb = mask + ((size_t)b * SEQ + q0) * SEQ;
    float* attnb = attn + ((size_t)b * SEQ + q0) * SEQ;

    // ---- Q tiles (2 x 16x64 fp32 -> bf16) in WMMA A-operand layout -------
    // A 16x32 bf16: lane(l&15)=M row; element e -> K = half*16 + e
    V16 qa[2][2];                 // [qtile][K-chunk]
#pragma unroll
    for (int qt = 0; qt < 2; ++qt)
#pragma unroll
        for (int c = 0; c < 2; ++c)
#pragma unroll
            for (int e = 0; e < 16; ++e) {
                const int kd = c * 32 + lhalf * 16 + e;
                qa[qt][c].e[e] = (__bf16)qb[(qt * 16 + lrow) * DIM + kd];
            }

    // ---- Phase 1: S = (Q K^T) * scale; each wave owns 256 keys -----------
    for (int i = 0; i < KCHUNK / 16; ++i) {
        const int key0 = wave * KCHUNK + i * 16;
        // B = K^T (64x16): lane(l&15)=key column; element e -> Kdim = half*16 + e
        V16 kbt[2];
#pragma unroll
        for (int c = 0; c < 2; ++c) {
            const float* rp = kb + (size_t)(key0 + lrow) * DIM + c * 32 + lhalf * 16;
#pragma unroll
            for (int e = 0; e < 16; ++e) kbt[c].e[e] = (__bf16)rp[e];
        }
#pragma unroll
        for (int qt = 0; qt < 2; ++qt) {
            V8F acc = {};
            acc.v = __builtin_amdgcn_wmma_f32_16x16x32_bf16(
                false, qa[qt][0].v, false, kbt[0].v, (short)0, acc.v, false, false);
            acc.v = __builtin_amdgcn_wmma_f32_16x16x32_bf16(
                false, qa[qt][1].v, false, kbt[1].v, (short)0, acc.v, false, false);
            // D layout: vgpr r -> row (r + 8*half), col = lane&15
#pragma unroll
            for (int r = 0; r < 8; ++r)
                sc[qt * 16 + r + 8 * lhalf][key0 + lrow] = acc.e[r] * SCALE;
        }
    }
    __syncthreads();

    // ---- Phase 2: mask + softmax, float4-vectorized (8 threads / row) ----
    {
        const int r = tid >> 3;          // query row 0..31
        const int t = tid & 7;
        const unsigned int* mrow = (const unsigned int*)(mb + (size_t)r * SEQ);
        const float NEG_INF = -__builtin_inff();

        float mx = NEG_INF;
        for (int jj = 0; jj < SEQ / 32; ++jj) {
            const int j = t * 4 + jj * 32;
            f4 s = *(const f4*)&sc[r][j];
            const unsigned int m = __builtin_nontemporal_load(mrow + (j >> 2));
            if (m & 0x000000ffu) s.x = NEG_INF;
            if (m & 0x0000ff00u) s.y = NEG_INF;
            if (m & 0x00ff0000u) s.z = NEG_INF;
            if (m & 0xff000000u) s.w = NEG_INF;
            *(f4*)&sc[r][j] = s;
            mx = fmaxf(mx, fmaxf(fmaxf(s.x, s.y), fmaxf(s.z, s.w)));
        }
#pragma unroll
        for (int o = 4; o; o >>= 1) mx = fmaxf(mx, __shfl_xor(mx, o, 32));

        float sum = 0.0f;
        for (int jj = 0; jj < SEQ / 32; ++jj) {
            const int j = t * 4 + jj * 32;
            f4 s = *(const f4*)&sc[r][j];
            s.x = __expf(s.x - mx); s.y = __expf(s.y - mx);
            s.z = __expf(s.z - mx); s.w = __expf(s.w - mx);
            *(f4*)&sc[r][j] = s;
            sum += (s.x + s.y) + (s.z + s.w);
        }
#pragma unroll
        for (int o = 4; o; o >>= 1) sum += __shfl_xor(sum, o, 32);
        const float inv = 1.0f / sum;

        for (int jj = 0; jj < SEQ / 32; ++jj) {
            const int j = t * 4 + jj * 32;
            f4 s = *(const f4*)&sc[r][j];
            s *= inv;
            *(f4*)&sc[r][j] = s;
            // 512MB streaming output: b128 NT stores, bypass L2 retention
            __builtin_nontemporal_store(s, (f4*)(attnb + (size_t)r * SEQ + j));
        }
    }
    __syncthreads();

    // ---- Phase 3: O = P V; each wave accumulates its 256-key slice -------
    V8F oacc[2][4] = {};
    const int kbase = wave * KCHUNK;
    for (int kt = 0; kt < KCHUNK; kt += 32) {
        const int kk = kbase + kt;
        // A = P tiles (16 queries x 32 keys) from LDS probs; K contiguous in
        // groups of 8 -> 4 x b128 LDS loads per tile.
        V16 pa[2];
#pragma unroll
        for (int qt = 0; qt < 2; ++qt)
#pragma unroll
            for (int gg = 0; gg < 2; ++gg) {
                const f4 lo = *(const f4*)&sc[qt * 16 + lrow][kk + gg * 16 + lhalf * 8];
                const f4 hi = *(const f4*)&sc[qt * 16 + lrow][kk + gg * 16 + lhalf * 8 + 4];
                pa[qt].e[gg * 8 + 0] = (__bf16)lo.x; pa[qt].e[gg * 8 + 1] = (__bf16)lo.y;
                pa[qt].e[gg * 8 + 2] = (__bf16)lo.z; pa[qt].e[gg * 8 + 3] = (__bf16)lo.w;
                pa[qt].e[gg * 8 + 4] = (__bf16)hi.x; pa[qt].e[gg * 8 + 5] = (__bf16)hi.y;
                pa[qt].e[gg * 8 + 6] = (__bf16)hi.z; pa[qt].e[gg * 8 + 7] = (__bf16)hi.w;
            }
#pragma unroll
        for (int g = 0; g < 4; ++g) {
            // B = V tile (32 keys x 16 dims), col = g*16 + (lane&15); shared
            // by both query tiles -> 2 WMMAs per staged tile.
            V16 vt;
#pragma unroll
            for (int e = 0; e < 16; ++e)
                vt.e[e] = (__bf16)vb[(size_t)(kk + ((e >> 3) << 4) + lhalf * 8 + (e & 7)) * DIM + g * 16 + lrow];
#pragma unroll
            for (int qt = 0; qt < 2; ++qt)
                oacc[qt][g].v = __builtin_amdgcn_wmma_f32_16x16x32_bf16(
                    false, pa[qt].v, false, vt.v, (short)0, oacc[qt][g].v, false, false);
        }
    }
    // Park each wave's 32x64 partial in its own (now dead) LDS columns.
#pragma unroll
    for (int qt = 0; qt < 2; ++qt)
#pragma unroll
        for (int g = 0; g < 4; ++g)
#pragma unroll
            for (int r = 0; r < 8; ++r)
                sc[qt * 16 + r + 8 * lhalf][wave * KCHUNK + g * 16 + lrow] = oacc[qt][g].e[r];
    __syncthreads();

    // ---- Final cross-wave reduction + coalesced b128 stores --------------
    {
        const int rr = tid >> 3;        // query row 0..31
        const int dd = (tid & 7) * 8;   // dim 0..63 in steps of 8
        f4 s0 = {0.f, 0.f, 0.f, 0.f}, s1 = {0.f, 0.f, 0.f, 0.f};
#pragma unroll
        for (int w = 0; w < NWAVES; ++w) {
            s0 += *(const f4*)&sc[rr][w * KCHUNK + dd];
            s1 += *(const f4*)&sc[rr][w * KCHUNK + dd + 4];
        }
        float* op = out + ((size_t)b * SEQ + q0 + rr) * DIM + dd;
        *(f4*)(op)     = s0;
        *(f4*)(op + 4) = s1;
    }
}

extern "C" void kernel_launch(void* const* d_in, const int* in_sizes, int n_in,
                              void* d_out, int out_size, void* d_ws, size_t ws_size,
                              hipStream_t stream) {
    const float* q = (const float*)d_in[0];
    const float* k = (const float*)d_in[1];
    const float* v = (const float*)d_in[2];
    const unsigned char* mask = (const unsigned char*)d_in[3];

    float* out  = (float*)d_out;                    // [B,S,D]
    float* attn = out + (size_t)BATCH * SEQ * DIM;  // [B,S,S]

    dim3 grid(SEQ / SQ, BATCH);   // 64 query tiles x 32 batches
    sdpa_wmma_kernel<<<grid, NTHREADS, 0, stream>>>(q, k, v, mask, out, attn);
}
